// KnnResidualGatedGCNModel_73933567034074
// MI455X (gfx1250) — matrix operand order repaired and checked by reading, not compiled, not run on hardware
//
#include <hip/hip_runtime.h>
#include <hip/hip_bf16.h>
#include <math.h>

typedef __attribute__((ext_vector_type(16))) _Float16 v16h;
typedef __attribute__((ext_vector_type(8)))  float    v8f;

#define Bn   16
#define Nn   1024
#define Kn   16
#define Hn   128
#define H2n  64
#define En   (Bn*Nn*Kn)    // 262144 edges
#define BNn  (Bn*Nn)       // 16384 nodes
#define WSZ  (Hn*Hn)       // 16384 elems per weight matrix

// ---------------------------------------------------------------------------
// Weight prep: f32 [128,128] (k-major, n minor) -> f16 WMMA B-fragment order.
// Fragment index f = ((ks*8 + nt)*32 + lane)*16 + elem, with
//   k = ks*32 + (lane>>4)*8 + (elem&7) + (elem>>3)*16   (B 32x16 f16: matches A table order)
//   n = (lane&15) + nt*16
// ---------------------------------------------------------------------------
__global__ void wprep_kernel(const float* __restrict__ w, _Float16* __restrict__ frag) {
    int f = blockIdx.x * blockDim.x + threadIdx.x;   // 16384 threads
    int elem =  f        & 15;
    int lane = (f >> 4)  & 31;
    int nt   = (f >> 9)  & 7;
    int ks   = (f >> 12) & 3;
    int kk = ks * 32 + (lane >> 4) * 8 + (elem & 7) + (elem >> 3) * 16;
    int n  = (lane & 15) + nt * 16;
    frag[f] = (_Float16)w[kk * Hn + n];
}

// ---------------------------------------------------------------------------
// KNN: one 128-thread block per (b,j). Per-thread sorted top-16 over a strided
// scan of the 1024 candidates, then 16 rounds of block-wide argmin merge.
// ---------------------------------------------------------------------------
__global__ void knn_kernel(const float* __restrict__ coord, int* __restrict__ src) {
    __shared__ float sd[128 * 16];
    __shared__ int   si[128 * 16];
    __shared__ float rd[128];
    __shared__ int   ri[128];
    __shared__ int   rt[128];
    int bj  = blockIdx.x;            // b*N + j
    int j   = bj & (Nn - 1);
    int tid = threadIdx.x;           // 0..127
    int rowBase = (bj >> 10) * Nn;   // b*N
    float cx = coord[bj * 2 + 0], cy = coord[bj * 2 + 1];

    float bd[16]; int bi[16];
#pragma unroll
    for (int s = 0; s < 16; ++s) { bd[s] = 3.0e38f; bi[s] = 0; }
    for (int i = tid; i < Nn; i += 128) {
        float dx = coord[(rowBase + i) * 2 + 0] - cx;
        float dy = coord[(rowBase + i) * 2 + 1] - cy;
        float d  = dx * dx + dy * dy + (i == j ? 1.0e9f : 0.0f);
        if (d < bd[15]) {
            bd[15] = d; bi[15] = i;
#pragma unroll
            for (int s = 15; s > 0; --s) {
                if (bd[s] < bd[s - 1]) {
                    float td = bd[s]; bd[s] = bd[s - 1]; bd[s - 1] = td;
                    int   ti = bi[s]; bi[s] = bi[s - 1]; bi[s - 1] = ti;
                }
            }
        }
    }
#pragma unroll
    for (int s = 0; s < 16; ++s) { sd[tid * 16 + s] = bd[s]; si[tid * 16 + s] = bi[s]; }
    __syncthreads();

    int p = 0;
    for (int k = 0; k < 16; ++k) {
        rd[tid] = (p < 16) ? sd[tid * 16 + p] : 3.0e38f;
        ri[tid] = (p < 16) ? si[tid * 16 + p] : 0;
        rt[tid] = tid;
        __syncthreads();
        for (int off = 64; off > 0; off >>= 1) {
            if (tid < off && rd[tid + off] < rd[tid]) {
                rd[tid] = rd[tid + off]; ri[tid] = ri[tid + off]; rt[tid] = rt[tid + off];
            }
            __syncthreads();
        }
        if (tid == 0) src[bj * Kn + k] = ri[0];
        int w = rt[0];
        __syncthreads();
        if (tid == w) p++;
    }
}

// ---------------------------------------------------------------------------
// h init: h = coord @ W_coord + E_node[x_nodes]
// ---------------------------------------------------------------------------
__global__ void inith_kernel(const float* __restrict__ coord, const int* __restrict__ nodes,
                             const float* __restrict__ Wc, const float* __restrict__ Enode,
                             float* __restrict__ h) {
    int idx = blockIdx.x * blockDim.x + threadIdx.x;   // BNn*Hn threads
    int v = idx >> 7, c = idx & 127;
    h[idx] = coord[v * 2] * Wc[c] + coord[v * 2 + 1] * Wc[Hn + c] + Enode[nodes[v] * Hn + c];
}

// ---------------------------------------------------------------------------
// e init: e = concat(ev*w_ev + b_ev, E_edge[et])
// ---------------------------------------------------------------------------
__global__ void inite_kernel(const int* __restrict__ xe, const float* __restrict__ xev,
                             const int* __restrict__ src, const float* __restrict__ wev,
                             const float* __restrict__ bev, const float* __restrict__ Eedge,
                             float* __restrict__ e) {
    int idx = blockIdx.x * blockDim.x + threadIdx.x;   // En*Hn threads
    int eI = idx >> 7, c = idx & 127;
    int b = eI >> 14;                // / (N*K)
    int j = (eI >> 4) & (Nn - 1);    // target
    int i = src[eI];                 // source
    int pos = (b * Nn + i) * Nn + j;
    if (c < H2n) e[idx] = xev[pos] * wev[c] + bev[c];
    else         e[idx] = Eedge[xe[pos] * H2n + (c - H2n)];
}

// ---------------------------------------------------------------------------
// WMMA GEMM: out[M,128] = act[M,128] @ W (f16 fragments, staged in LDS).
// Block = 256 threads = 8 waves; each wave owns a 16-row strip, 8 column
// tiles, K-loop of 4 -> 32 v_wmma_f32_16x16x32_f16 per wave. No divergence
// (EXEC all-1s for WMMA).
// ---------------------------------------------------------------------------
__global__ void __launch_bounds__(256) gemm_kernel(const float* __restrict__ act,
                                                   const _Float16* __restrict__ wfrag,
                                                   float* __restrict__ out, int relu) {
    __shared__ __align__(32) _Float16 wl[WSZ];
    {   // cooperative 32KB stage of the whole weight matrix
        const uint4* s = (const uint4*)wfrag;
        uint4*       d = (uint4*)wl;
        for (int t = threadIdx.x; t < (WSZ * 2) / 16; t += 256) d[t] = s[t];
    }
    __syncthreads();

    int wave = threadIdx.x >> 5, lane = threadIdx.x & 31;
    int lh = lane >> 4, ln = lane & 15;
    int rowBase = blockIdx.x * 128 + wave * 16;
    int row = rowBase + ln;                       // A-matrix M = lane&15

    v8f acc[8] = {};
#pragma unroll
    for (int ks = 0; ks < 4; ++ks) {
        // A 16x32 f16 layout: elems 0-7 -> K = ks*32 + lh*8 + i,
        //                     elems 8-15 -> K = ks*32 + 16 + lh*8 + i
        const float* ap = act + row * Hn + ks * 32 + lh * 8;
        __builtin_prefetch(ap + 32, 0, 1);        // global_prefetch_b8 for next K step
        float4 x0 = *(const float4*)(ap);
        float4 x1 = *(const float4*)(ap + 4);
        float4 x2 = *(const float4*)(ap + 16);
        float4 x3 = *(const float4*)(ap + 20);
        v16h a;
        a[0]=(_Float16)x0.x; a[1]=(_Float16)x0.y; a[2]=(_Float16)x0.z; a[3]=(_Float16)x0.w;
        a[4]=(_Float16)x1.x; a[5]=(_Float16)x1.y; a[6]=(_Float16)x1.z; a[7]=(_Float16)x1.w;
        a[8]=(_Float16)x2.x; a[9]=(_Float16)x2.y; a[10]=(_Float16)x2.z; a[11]=(_Float16)x2.w;
        a[12]=(_Float16)x3.x; a[13]=(_Float16)x3.y; a[14]=(_Float16)x3.z; a[15]=(_Float16)x3.w;
#pragma unroll
        for (int nt = 0; nt < 8; ++nt) {
            const v16h* bp = (const v16h*)&wl[((ks * 8 + nt) * 32 + lane) * 16];
            acc[nt] = __builtin_amdgcn_wmma_f32_16x16x32_f16(
                false, a, false, *bp, (short)0, acc[nt], false, false);
        }
    }
    // C/D layout: VGPR r, lane -> (M = r + 8*lh, N = ln + nt*16)
#pragma unroll
    for (int nt = 0; nt < 8; ++nt) {
#pragma unroll
        for (int r = 0; r < 8; ++r) {
            float v = acc[nt][r];
            if (relu) v = fmaxf(v, 0.0f);
            out[(rowBase + r + 8 * lh) * Hn + ln + nt * 16] = v;
        }
    }
}

// ---------------------------------------------------------------------------
// e_new = e_old + relu(eU + hV[src] + hWt[tgt])   (in-place on etmp = eU)
// ---------------------------------------------------------------------------
__global__ void ecomb_kernel(const float* __restrict__ eold, float* __restrict__ etmp,
                             const float* __restrict__ hV, const float* __restrict__ hWt,
                             const int* __restrict__ src) {
    int idx = blockIdx.x * blockDim.x + threadIdx.x;   // En*Hn threads
    int eI = idx >> 7, c = idx & 127;
    int s = src[eI];
    int t = eI >> 4;                                   // tgt flat node (edges contiguous by tgt)
    float v = etmp[idx] + hV[s * Hn + c] + hWt[t * Hn + c];
    etmp[idx] = eold[idx] + fmaxf(v, 0.0f);
}

// ---------------------------------------------------------------------------
// h += relu(hA + segsum(sigmoid(e)*hBw[src]) / (segsum(sigmoid(e)) + eps))
// Contiguous K=16 reduction per target node -> no atomics.
// ---------------------------------------------------------------------------
__global__ void nupd_kernel(float* __restrict__ h, const float* __restrict__ hA,
                            const float* __restrict__ hBw, const float* __restrict__ enew,
                            const int* __restrict__ src) {
    int idx = blockIdx.x * blockDim.x + threadIdx.x;   // BNn*Hn threads
    int t = idx >> 7, c = idx & 127;
    float num = 0.0f, den = 0.0f;
#pragma unroll
    for (int k = 0; k < Kn; ++k) {
        int eI = t * Kn + k;
        float g = 1.0f / (1.0f + __expf(-enew[eI * Hn + c]));
        num += g * hBw[src[eI] * Hn + c];
        den += g;
    }
    h[idx] += fmaxf(hA[idx] + num / (den + 1e-20f), 0.0f);
}

// ---------------------------------------------------------------------------
// y[e,2] = relu_hidden[e,:] @ Wm2[128,2]
// ---------------------------------------------------------------------------
__global__ void eout_kernel(const float* __restrict__ r, const float* __restrict__ Wm2,
                            float* __restrict__ y) {
    int e = blockIdx.x * blockDim.x + threadIdx.x;     // En threads
    float a0 = 0.0f, a1 = 0.0f;
    const float* rp = r + e * Hn;
#pragma unroll 4
    for (int c = 0; c < Hn; ++c) {
        float rv = rp[c];
        a0 += rv * Wm2[c * 2 + 0];
        a1 += rv * Wm2[c * 2 + 1];
    }
    y[e * 2 + 0] = a0;
    y[e * 2 + 1] = a1;
}

// ---------------------------------------------------------------------------
// dense init: [..., 0] = 0, [..., 1] = logit_noedge
// ---------------------------------------------------------------------------
__global__ void oinit_kernel(float* __restrict__ out, const float* __restrict__ lnv) {
    int i = blockIdx.x * blockDim.x + threadIdx.x;     // B*N*N threads
    ((float2*)out)[i] = make_float2(0.0f, lnv[0]);
}

__global__ void oscat_kernel(float* __restrict__ out, const float* __restrict__ y,
                             const int* __restrict__ src) {
    int e = blockIdx.x * blockDim.x + threadIdx.x;     // En threads
    int b = e >> 14;
    int j = (e >> 4) & (Nn - 1);
    int i = src[e];
    int pos = (b * Nn + i) * Nn + j;
    out[pos * 2 + 0] = y[e * 2 + 0];
    out[pos * 2 + 1] = y[e * 2 + 1];
}

// ---------------------------------------------------------------------------
extern "C" void kernel_launch(void* const* d_in, const int* in_sizes, int n_in,
                              void* d_out, int out_size, void* d_ws, size_t ws_size,
                              hipStream_t stream) {
    const int*   xe    = (const int*)d_in[0];
    const float* xev   = (const float*)d_in[1];
    const int*   xn    = (const int*)d_in[2];
    const float* coord = (const float*)d_in[3];
    const float* Wc    = (const float*)d_in[4];
    const float* Enode = (const float*)d_in[5];
    const float* wev   = (const float*)d_in[6];
    const float* bev   = (const float*)d_in[7];
    const float* Eedge = (const float*)d_in[8];
    const float* U     = (const float*)d_in[9];
    const float* V     = (const float*)d_in[10];
    const float* Wt    = (const float*)d_in[11];
    const float* A     = (const float*)d_in[12];
    const float* Bw    = (const float*)d_in[13];
    const float* Wm1   = (const float*)d_in[14];
    const float* Wm2   = (const float*)d_in[15];
    const float* lnv   = (const float*)d_in[16];

    // workspace carve-up
    char* ws = (char*)d_ws;
    size_t off = 0;
    auto carve = [&](size_t bytes) {
        void* p = ws + off;
        off = (off + bytes + 255) & ~(size_t)255;
        return p;
    };
    _Float16* frag = (_Float16*)carve((size_t)16 * WSZ * sizeof(_Float16)); // 512 KB
    int*   src  = (int*)carve((size_t)En * sizeof(int));
    float* h    = (float*)carve((size_t)BNn * Hn * sizeof(float));
    float* hV   = (float*)carve((size_t)BNn * Hn * sizeof(float));
    float* hWt  = (float*)carve((size_t)BNn * Hn * sizeof(float));
    float* hBw  = (float*)carve((size_t)BNn * Hn * sizeof(float));
    float* hA   = (float*)carve((size_t)BNn * Hn * sizeof(float));
    float* eA   = (float*)carve((size_t)En * Hn * sizeof(float));
    float* eB   = (float*)carve((size_t)En * Hn * sizeof(float));
    float* yE   = (float*)carve((size_t)En * 2 * sizeof(float));

    // fragment slots: U -> 0..2, V -> 3..5, Wt -> 6..8, A -> 9..11, Bw -> 12..14, Wm1 -> 15
    for (int l = 0; l < 3; ++l) {
        wprep_kernel<<<64, 256, 0, stream>>>(U  + (size_t)l * WSZ, frag + (size_t)(0  + l) * WSZ);
        wprep_kernel<<<64, 256, 0, stream>>>(V  + (size_t)l * WSZ, frag + (size_t)(3  + l) * WSZ);
        wprep_kernel<<<64, 256, 0, stream>>>(Wt + (size_t)l * WSZ, frag + (size_t)(6  + l) * WSZ);
        wprep_kernel<<<64, 256, 0, stream>>>(A  + (size_t)l * WSZ, frag + (size_t)(9  + l) * WSZ);
        wprep_kernel<<<64, 256, 0, stream>>>(Bw + (size_t)l * WSZ, frag + (size_t)(12 + l) * WSZ);
    }
    wprep_kernel<<<64, 256, 0, stream>>>(Wm1, frag + (size_t)15 * WSZ);

    knn_kernel<<<BNn, 128, 0, stream>>>(coord, src);
    inith_kernel<<<(BNn * Hn) / 256, 256, 0, stream>>>(coord, xn, Wc, Enode, h);
    inite_kernel<<<(En * Hn) / 256, 256, 0, stream>>>(xe, xev, src, wev, bev, Eedge, eA);

    float* ecur = eA;
    float* etmp = eB;
    for (int l = 0; l < 3; ++l) {
        // per-node transforms (4 small GEMMs instead of [E,H] GEMMs)
        gemm_kernel<<<BNn / 128, 256, 0, stream>>>(h, frag + (size_t)(3  + l) * WSZ, hV,  0);
        gemm_kernel<<<BNn / 128, 256, 0, stream>>>(h, frag + (size_t)(6  + l) * WSZ, hWt, 0);
        gemm_kernel<<<BNn / 128, 256, 0, stream>>>(h, frag + (size_t)(12 + l) * WSZ, hBw, 0);
        gemm_kernel<<<BNn / 128, 256, 0, stream>>>(h, frag + (size_t)(9  + l) * WSZ, hA,  0);
        // edge GEMM: eU = e @ U
        gemm_kernel<<<En / 128, 256, 0, stream>>>(ecur, frag + (size_t)(0 + l) * WSZ, etmp, 0);
        // e_new = e + relu(eU + hV[src] + hWt[tgt])
        ecomb_kernel<<<(En * Hn) / 256, 256, 0, stream>>>(ecur, etmp, hV, hWt, src);
        { float* t = ecur; ecur = etmp; etmp = t; }
        // h update with contiguous-K segment sums of sigmoid(e_new)
        nupd_kernel<<<(BNn * Hn) / 256, 256, 0, stream>>>(h, hA, hBw, ecur, src);
    }

    // MLP head: hidden = relu(e @ Wm1), y = hidden @ Wm2
    gemm_kernel<<<En / 128, 256, 0, stream>>>(ecur, frag + (size_t)15 * WSZ, etmp, 1);
    eout_kernel<<<En / 256, 256, 0, stream>>>(etmp, Wm2, yE);

    // dense output
    oinit_kernel<<<(Bn * Nn * Nn) / 256, 256, 0, stream>>>((float*)d_out, lnv);
    oscat_kernel<<<En / 256, 256, 0, stream>>>((float*)d_out, yE, src);
}
